// GMM_81329500717562
// MI455X (gfx1250) — matrix operand order repaired and checked by reading, not compiled, not run on hardware
//
#include <hip/hip_runtime.h>
#include <math.h>

typedef __attribute__((ext_vector_type(16))) _Float16 v16h;
typedef __attribute__((ext_vector_type(8)))  float    v8f;

#define DD   32          // feature dim
#define KK   16          // clusters
#define SS   136         // upper-tri pairs
#define MM   100         // interpolation steps
#define NCOL (SS*MM)     // 13600 = 16*850
#define NROW 2048        // 256*8   = 16*128
#define COLT (NCOL/16)   // 850 column tiles
#define ROWT (NROW/16)   // 128 row tiles

#define LOG2E 1.4426950408889634f

// ---------------- Kernel P: softmax(pi) + mu passthrough -------------------
__global__ void gmm_softmax_mu(const float* __restrict__ pi,
                               const float* __restrict__ mu,
                               float* __restrict__ out,
                               float* __restrict__ logpi) {
  __shared__ float buf[SS];
  __shared__ float red[2];
  const int tid = threadIdx.x;
  if (tid < SS) buf[tid] = pi[tid];
  __syncthreads();
  if (tid == 0) {
    float mx = buf[0];
    for (int i = 1; i < SS; i++) mx = fmaxf(mx, buf[i]);
    float sm = 0.f;
    for (int i = 0; i < SS; i++) sm += expf(buf[i] - mx);
    red[0] = mx; red[1] = sm;
  }
  __syncthreads();
  if (tid < SS) {
    float sp = expf(buf[tid] - red[0]) / red[1];
    out[tid]   = sp;                      // sm_pi -> output slot 0..135
    logpi[tid] = logf(sp + 1e-30f);       // matches reference log(sm_pi+1e-30)
  }
  out[SS + tid]       = mu[tid];          // mu passthrough (512 floats)
  out[SS + 256 + tid] = mu[256 + tid];
}

// ------- Kernel Q: build split-f16 B matrix (WMMA lane layout) + c[s,m] ----
// B tile layout for v_wmma_f32_16x16x32_f16 (wave32):
//   lane = N (0..15) | N+16 ; halves e=0..15 hold K = (lane<16 ? e : 16+e)
// cvec is pre-scaled by log2(e) so the main loop feeds v_exp_f32 directly.
__global__ void gmm_build_B(const float* __restrict__ mu,
                            const float* __restrict__ logpi,
                            _Float16* __restrict__ bhi,
                            _Float16* __restrict__ blo,
                            float* __restrict__ cvec) {
  const int n = blockIdx.x * blockDim.x + threadIdx.x;
  if (n >= NCOL) return;
  const int s = n / MM;
  const int m = n - s * MM;
  const float w = (float)m * (1.0f / MM);
  // decode (a,b): row-major enumeration of upper-tri (incl diag) of 16x16
  int a = 0, rem = s;
  while (rem >= (KK - a)) { rem -= (KK - a); a++; }
  const int b = a + rem;
  const int t  = n >> 4;
  const int cl = n & 15;
  float mw2 = 0.f;
  for (int d = 0; d < DD; d++) {
    float v = mu[d * KK + a] * w + mu[d * KK + b] * (1.f - w);
    mw2 += v * v;
    _Float16 h = (_Float16)v;
    _Float16 l = (_Float16)(v - (float)h);      // split-f16 residual
    const int lane = cl + ((d >= 16) ? 16 : 0);
    const int e    = d & 15;
    const int idx  = (t * 32 + lane) * 16 + e;
    bhi[idx] = h;
    blo[idx] = l;
  }
  cvec[n] = (logpi[s] - 0.5f * mw2) * LOG2E;    // base-2 exponent domain
}

// ---------------- Main kernel: WMMA GEMM + streaming sum-exp ---------------
__global__ void __launch_bounds__(256) gmm_main(
    const float*    __restrict__ z,
    const _Float16* __restrict__ bhi,
    const _Float16* __restrict__ blo,
    const float*    __restrict__ cvec,
    float*          __restrict__ rowlp) {
  __shared__ float z2s[16];
  __shared__ float wsum[8][16];
  const int i    = blockIdx.x;       // row tile (16 rows of z)
  const int tid  = threadIdx.x;
  const int lane = tid & 31;
  const int wv   = tid >> 5;         // 8 waves per block

  // |z|^2 per row of this tile (one-time, 16 threads, natural units)
  if (tid < 16) {
    const float* zr = z + (size_t)(i * 16 + tid) * DD;
    float s = 0.f;
    for (int d = 0; d < DD; d++) s += zr[d] * zr[d];
    z2s[tid] = s;
  }

  // A fragment: lane c (=lane&15) holds row 16*i+c, PRE-SCALED by log2(e)
  // so cross arrives in base-2 exponent units (saves a mul before v_exp_f32).
  // A layout (matches B packing): lanes 0-15 -> elems 0..15 = K0..15;
  //               lanes 16-31 -> K16..31.
  const int c = lane & 15;
  const int h = lane >> 4;
  const float* zrow = z + (size_t)(i * 16 + c) * DD;
  v16h ahi, alo;
#pragma unroll
  for (int e = 0; e < 8; e++) {
    float x0 = zrow[h * 16 + e]     * LOG2E;
    float x1 = zrow[h * 16 + 8 + e] * LOG2E;
    _Float16 h0 = (_Float16)x0;
    _Float16 h1 = (_Float16)x1;
    ahi[e]     = h0;  alo[e]     = (_Float16)(x0 - (float)h0);
    ahi[e + 8] = h1;  alo[e + 8] = (_Float16)(x1 - (float)h1);
  }

  float acc[8];
#pragma unroll
  for (int v = 0; v < 8; v++) acc[v] = 0.f;

  // Register double-buffer: loads for tile t+8 are issued before the WMMA
  // chain on tile t, so s_wait_loadcnt has a full iteration to hide.
  size_t bidx = (size_t)(wv * 32 + lane) * 16;
  v16h  bh = *(const v16h*)(bhi + bidx);
  v16h  bl = *(const v16h*)(blo + bidx);
  float cc = cvec[wv * 16 + c];

  for (int t = wv; t < COLT; t += 8) {
    const int tn = (t + 8 < COLT) ? (t + 8) : t;     // clamp: tail reloads self
    const size_t nidx = (size_t)(tn * 32 + lane) * 16;
    v16h  bh_n = *(const v16h*)(bhi + nidx);
    v16h  bl_n = *(const v16h*)(blo + nidx);
    float cc_n = cvec[tn * 16 + c];
    {   // unconditional clamped prefetch two tiles ahead (speculative ok)
      const int tp = (t + 16 < COLT) ? (t + 16) : t;
      const size_t pidx = (size_t)(tp * 32 + lane) * 16;
      __builtin_prefetch(bhi + pidx, 0, 1);
      __builtin_prefetch(blo + pidx, 0, 1);
    }
    // split-f16 product: ahi*blo + alo*bhi + ahi*bhi  (near-f32 accuracy)
    v8f ctile = {};
    ctile = __builtin_amdgcn_wmma_f32_16x16x32_f16(false, ahi, false, bl,
                                                   (short)0, ctile, false, false);
    ctile = __builtin_amdgcn_wmma_f32_16x16x32_f16(false, alo, false, bh,
                                                   (short)0, ctile, false, false);
    ctile = __builtin_amdgcn_wmma_f32_16x16x32_f16(false, ahi, false, bh,
                                                   (short)0, ctile, false, false);
#pragma unroll
    for (int v = 0; v < 8; v++)
      acc[v] += __builtin_amdgcn_exp2f(ctile[v] + cc);   // raw v_exp_f32
    bh = bh_n; bl = bl_n; cc = cc_n;
  }

  // C layout: lanes 0-15 hold rows 0..7 (VGPR v -> row v), lanes 16-31 rows 8..15.
  // Reduce over the 16 columns held within each half-wave.
#pragma unroll
  for (int v = 0; v < 8; v++) {
    float a = acc[v];
    a += __shfl_xor(a, 1);
    a += __shfl_xor(a, 2);
    a += __shfl_xor(a, 4);
    a += __shfl_xor(a, 8);
    acc[v] = a;
  }
  if (c == 0) {                       // lane 0 -> rows 0..7, lane 16 -> rows 8..15
#pragma unroll
    for (int v = 0; v < 8; v++) wsum[wv][h * 8 + v] = acc[v];
  }
  __syncthreads();

  if (tid < 16) {
    float tot = 0.f;
    for (int w = 0; w < 8; w++) tot += wsum[w][tid];   // fixed order: deterministic
    // C0 = -0.5*D*log(2*pi) - log(M)   (natural-log units)
    const float C0 = -34.011203248537616f;
    rowlp[i * 16 + tid] = C0 + logf(tot) - 0.5f * z2s[tid];
  }
}

// ---------------- Final deterministic mean over 2048 rows ------------------
__global__ void gmm_reduce(const float* __restrict__ rowlp,
                           float* __restrict__ out) {
  __shared__ float sh[256];
  const int tid = threadIdx.x;
  float s = 0.f;
  for (int j = 0; j < 8; j++) s += rowlp[tid + 256 * j];
  sh[tid] = s;
  __syncthreads();
  for (int off = 128; off > 0; off >>= 1) {
    if (tid < off) sh[tid] += sh[tid + off];
    __syncthreads();
  }
  if (tid == 0) out[SS + 512] = sh[0] * (1.0f / (float)NROW);
}

extern "C" void kernel_launch(void* const* d_in, const int* in_sizes, int n_in,
                              void* d_out, int out_size, void* d_ws, size_t ws_size,
                              hipStream_t stream) {
  const float* z  = (const float*)d_in[0];   // [256,8,32]
  const float* pi = (const float*)d_in[1];   // [1,136]
  const float* mu = (const float*)d_in[2];   // [32,16]
  float* out = (float*)d_out;                // 136 + 512 + 1 = 649 floats

  char* ws = (char*)d_ws;
  const size_t BHALVES = (size_t)NCOL * DD * sizeof(_Float16);   // 870400 B
  float*    logpi = (float*)(ws);                                //   544 B
  _Float16* bhi   = (_Float16*)(ws + 1024);
  _Float16* blo   = (_Float16*)(ws + 1024 + BHALVES);
  float*    cvec  = (float*)(ws + 1024 + 2 * BHALVES);           // 54400 B
  float*    rowlp = (float*)(ws + 1024 + 2 * BHALVES + 57344);   //  8192 B

  gmm_softmax_mu<<<1, 256, 0, stream>>>(pi, mu, out, logpi);
  gmm_build_B<<<(NCOL + 255) / 256, 256, 0, stream>>>(mu, logpi, bhi, blo, cvec);
  gmm_main<<<ROWT, 256, 0, stream>>>(z, bhi, blo, cvec, rowlp);
  gmm_reduce<<<1, 256, 0, stream>>>(rowlp, out);
}
